// MultiHeadSelfAttention_56899726737486
// MI455X (gfx1250) — compile-verified, matrix operand order checked by reading, not run on hardware
//
#include <hip/hip_runtime.h>
#include <hip/hip_bf16.h>

typedef __bf16 bf16;
typedef __attribute__((ext_vector_type(16))) __bf16 v16bf;
typedef __attribute__((ext_vector_type(8)))  float  v8f;

#define D_MODEL   1024
#define NUM_HEADS 16
#define D_K       64
#define SEQ       2048
#define BATCH     4
#define M_TOTAL   (BATCH * SEQ)

// ---------- helpers ----------

__device__ __forceinline__ bf16 to_bf16(float f) {
  unsigned u = __builtin_bit_cast(unsigned, f);
  u += 0x7FFFu + ((u >> 16) & 1u);          // round-to-nearest-even
  unsigned short h = (unsigned short)(u >> 16);
  return __builtin_bit_cast(bf16, h);
}

// Load one 16x16x32 bf16 WMMA fragment (A or B operand) for this lane.
// p points at the lane's row base + k0 + (lane<16 ? 0 : 8).
// Layout per ISA: lane<16 holds K {0..7,16..23}, lane>=16 holds K {8..15,24..31}.
__device__ __forceinline__ v16bf load_frag(const bf16* p) {
  union { v16bf v; uint4 u[2]; } f;
  f.u[0] = *reinterpret_cast<const uint4*>(p);
  f.u[1] = *reinterpret_cast<const uint4*>(p + 16);
  return f.v;
}

// ---------- f32 -> bf16 conversion (vectorized x4) ----------

__global__ __launch_bounds__(256)
void cvt_f32_to_bf16(const float* __restrict__ in, bf16* __restrict__ out, int n4) {
  int i = blockIdx.x * 256 + threadIdx.x;
  if (i >= n4) return;
  float4 f = reinterpret_cast<const float4*>(in)[i];
  union { bf16 b[4]; ushort4 u; } o;
  o.b[0] = to_bf16(f.x); o.b[1] = to_bf16(f.y);
  o.b[2] = to_bf16(f.z); o.b[3] = to_bf16(f.w);
  reinterpret_cast<ushort4*>(out)[i] = o.u;
}

// ---------- bf16 WMMA GEMM: C[M,N] = A[M,K] * W[N,K]^T ----------
// Each wave owns a 32x64 C tile: 2 A fragments x 4 B fragments -> 8 WMMAs/k-step.
// MODE 0: fused RoPE epilogue, write bf16 [B,H,S,D_K]   (Q and K paths)
// MODE 1: write bf16 transposed [B,H,D_K,S]             (V path, for PV B-operand)
// MODE 2: write f32 [M, D_MODEL] directly               (output projection)
template <int MODE>
__global__ __launch_bounds__(128)
void gemm16_kernel(const bf16* __restrict__ A, const bf16* __restrict__ W,
                   void* __restrict__ OutP, const int* __restrict__ tp) {
  const int lane   = threadIdx.x & 31;
  const int wave   = threadIdx.x >> 5;
  const int lane15 = lane & 15;
  const int hiOff  = (lane < 16) ? 0 : 8;     // K-group split for A/B fragments
  const int hi8    = hiOff;                   // M/row split for C fragments
  const int m0 = (blockIdx.x * 4 + wave) * 32;
  const int n0 = blockIdx.y * 64;

  const bf16* aRow0 = A + (size_t)(m0 + lane15) * D_MODEL;
  const bf16* aRow1 = A + (size_t)(m0 + 16 + lane15) * D_MODEL;
  const bf16* wRow0 = W + (size_t)(n0 + lane15) * D_MODEL;

  v8f acc[2][4] = {};
  for (int k0 = 0; k0 < D_MODEL; k0 += 32) {
    __builtin_prefetch(aRow0 + k0 + 256, 0, 0);
    v16bf aF0 = load_frag(aRow0 + k0 + hiOff);
    v16bf aF1 = load_frag(aRow1 + k0 + hiOff);
#pragma unroll
    for (int t = 0; t < 4; ++t) {
      v16bf bF = load_frag(wRow0 + (size_t)t * 16 * D_MODEL + k0 + hiOff);
      acc[0][t] = __builtin_amdgcn_wmma_f32_16x16x32_bf16(
          false, aF0, false, bF, (short)0, acc[0][t], false, false);
      acc[1][t] = __builtin_amdgcn_wmma_f32_16x16x32_bf16(
          false, aF1, false, bF, (short)0, acc[1][t], false, false);
    }
  }

#pragma unroll
  for (int mi = 0; mi < 2; ++mi) {
    if constexpr (MODE == 0) {
      // RoPE fused: pair partner (even<->odd feature) lives in lane^1.
      bf16* out = (bf16*)OutP;
      const int h = n0 >> 6;                       // one head per 64-wide N tile
#pragma unroll
      for (int t = 0; t < 4; ++t) {
        const int d   = t * 16 + lane15;           // feature within head
        const int j   = d >> 1;
        const bool odd = (d & 1) != 0;
        const float invf = __expf(-(float)j * (9.210340371976184f / 32.0f)); // 10000^(-j/32)
#pragma unroll
        for (int r = 0; r < 8; ++r) {
          const int row = m0 + mi * 16 + r + hi8;
          const int bb  = row >> 11;
          const int s   = row & (SEQ - 1);
          float sn, cs;
          __sincosf((float)tp[s] * invf, &sn, &cs);
          float val   = acc[mi][t][r];
          float other = __shfl_xor(val, 1, 32);
          float e  = odd ? other : val;
          float od = odd ? val   : other;
          float res = odd ? (e * sn + od * cs) : (e * cs - od * sn);
          out[((size_t)((bb * NUM_HEADS + h) * SEQ + s) << 6) + d] = to_bf16(res);
        }
      }
    } else if constexpr (MODE == 1) {
      bf16* out = (bf16*)OutP;
      const int h = n0 >> 6;
#pragma unroll
      for (int t = 0; t < 4; ++t) {
        const int d = t * 16 + lane15;
#pragma unroll
        for (int r = 0; r < 8; ++r) {
          const int row = m0 + mi * 16 + r + hi8;
          const int bb  = row >> 11;
          const int s   = row & (SEQ - 1);
          out[((size_t)((bb * NUM_HEADS + h) * D_K + d) << 11) + s] = to_bf16(acc[mi][t][r]);
        }
      }
    } else {
      (void)tp;
      float* out = (float*)OutP;
#pragma unroll
      for (int t = 0; t < 4; ++t)
#pragma unroll
        for (int r = 0; r < 8; ++r)
          out[(size_t)(m0 + mi * 16 + r + hi8) * D_MODEL + n0 + t * 16 + lane15] =
              acc[mi][t][r];
    }
  }
}

// ---------- flash attention ----------
// One block = 4 waves = 64 q-rows of one (b,h). K/V chunks (32 keys) are staged
// into LDS once per block via async global->LDS (ASYNCcnt, double-buffered),
// then all 4 waves consume them as ds_load_b128 WMMA fragments.

__global__ __launch_bounds__(128)
void attn_kernel(const bf16* __restrict__ Q, const bf16* __restrict__ K,
                 const bf16* __restrict__ VT, bf16* __restrict__ Ob) {
  __shared__ bf16 kTile[2][32][64];   // [buf][key token][d]        8 KB
  __shared__ bf16 vTile[2][64][32];   // [buf][d][key token]        8 KB
  __shared__ bf16 pLds[4][16][32];    // per-wave probs staging     4 KB

  const int lane   = threadIdx.x & 31;
  const int wave   = threadIdx.x >> 5;
  const int lane15 = lane & 15;
  const int hiOff  = (lane < 16) ? 0 : 8;
  const int hi8    = hiOff;
  const int q0     = (blockIdx.x * 4 + wave) * 16;
  const int h  = blockIdx.y;
  const int b  = blockIdx.z;
  const size_t headOff = (size_t)(b * NUM_HEADS + h) * SEQ * D_K;
  const bf16* qh = Q  + headOff;
  const bf16* kh = K  + headOff;
  const bf16* vh = VT + headOff;    // [D_K][SEQ]

  // Cooperative async stage of one 32-key chunk: 128 threads x 2 x (K+V) 16B txfers.
  auto stage = [&](int buf, int kc) {
#pragma unroll
    for (int ii = 0; ii < 2; ++ii) {
      const int i = threadIdx.x + ii * 128;            // 0..255
      const int krow = i >> 3, kseg = i & 7;           // K: 32 rows x 8 segs
      unsigned kOff = (unsigned)((((kc + krow) * D_K) + kseg * 8) * 2);
      unsigned kLds = (unsigned)(size_t)&kTile[buf][krow][kseg * 8];
      asm volatile("global_load_async_to_lds_b128 %0, %1, %2"
                   :: "v"(kLds), "v"(kOff), "s"(kh) : "memory");
      const int vrow = i >> 2, vseg = i & 3;           // V: 64 rows x 4 segs
      unsigned vOff = (unsigned)(((vrow * SEQ) + kc + vseg * 8) * 2);
      unsigned vLds = (unsigned)(size_t)&vTile[buf][vrow][vseg * 8];
      asm volatile("global_load_async_to_lds_b128 %0, %1, %2"
                   :: "v"(vLds), "v"(vOff), "s"(vh) : "memory");
    }
  };

  const bf16* qRow = qh + (size_t)(q0 + lane15) * D_K;
  v16bf aQ0 = load_frag(qRow + hiOff);        // d = 0..31
  v16bf aQ1 = load_frag(qRow + 32 + hiOff);   // d = 32..63

  v8f o[4] = {};
  float mSt[8], lSt[8];
#pragma unroll
  for (int r = 0; r < 8; ++r) { mSt[r] = -3.0e38f; lSt[r] = 0.0f; }

  const int kcEnd = blockIdx.x * 64 + 63;     // block-max causal bound (wave 3)
  stage(0, 0);                                // prologue: fill buffer 0

  for (int kc = 0; kc <= kcEnd; kc += 32) {
    const int cur  = (kc >> 5) & 1;
    const bool more = (kc + 32 <= kcEnd);
    if (more) stage(cur ^ 1, kc + 32);        // issue-ahead into other buffer
    if (more) asm volatile("s_wait_asynccnt 0x4" ::: "memory");  // cur chunk done
    else      asm volatile("s_wait_asynccnt 0x0" ::: "memory");
    __syncthreads();

    if (kc <= q0 + 15) {                      // wave-uniform causal guard
      // ---- scores S = Q*K^T / sqrt(d_k) over a 16x32 key chunk (from LDS) ----
      v8f sc[2];
#pragma unroll
      for (int ns = 0; ns < 2; ++ns) {
        const bf16* kRow = &kTile[cur][ns * 16 + lane15][0];
        v16bf b0 = load_frag(kRow + hiOff);
        v16bf b1 = load_frag(kRow + 32 + hiOff);
        v8f c = {};
        c = __builtin_amdgcn_wmma_f32_16x16x32_bf16(false, aQ0, false, b0, (short)0, c, false, false);
        c = __builtin_amdgcn_wmma_f32_16x16x32_bf16(false, aQ1, false, b1, (short)0, c, false, false);
        sc[ns] = c;
      }
      // ---- scale + causal mask + row max ----
      float mx[8];
#pragma unroll
      for (int r = 0; r < 8; ++r) mx[r] = -3.0e38f;
#pragma unroll
      for (int ns = 0; ns < 2; ++ns) {
        const int col = kc + ns * 16 + lane15;
#pragma unroll
        for (int r = 0; r < 8; ++r) {
          const int row = q0 + r + hi8;
          float s = sc[ns][r] * 0.125f;           // 1/sqrt(64)
          if (col > row) s = -1.0e9f;
          sc[ns][r] = s;
          mx[r] = fmaxf(mx[r], s);
        }
      }
#pragma unroll
      for (int r = 0; r < 8; ++r)
#pragma unroll
        for (int msk = 1; msk <= 8; msk <<= 1)
          mx[r] = fmaxf(mx[r], __shfl_xor(mx[r], msk, 32));
      // ---- online softmax update ----
      float alpha[8], rsum[8];
#pragma unroll
      for (int r = 0; r < 8; ++r) {
        float mn = fmaxf(mSt[r], mx[r]);
        alpha[r] = __expf(mSt[r] - mn);
        mSt[r] = mn;
        rsum[r] = 0.0f;
      }
#pragma unroll
      for (int ns = 0; ns < 2; ++ns)
#pragma unroll
        for (int r = 0; r < 8; ++r) {
          float p = __expf(sc[ns][r] - mSt[r]);
          rsum[r] += p;
          pLds[wave][r + hi8][ns * 16 + lane15] = to_bf16(p);
        }
#pragma unroll
      for (int r = 0; r < 8; ++r) {
#pragma unroll
        for (int msk = 1; msk <= 8; msk <<= 1)
          rsum[r] += __shfl_xor(rsum[r], msk, 32);
        lSt[r] = lSt[r] * alpha[r] + rsum[r];
      }
#pragma unroll
      for (int t = 0; t < 4; ++t)
#pragma unroll
        for (int r = 0; r < 8; ++r) o[t][r] *= alpha[r];

      // ---- O += P * V  (P via LDS relayout, V fragments from LDS) ----
      __builtin_amdgcn_wave_barrier();
      v16bf pF = load_frag(&pLds[wave][lane15][0] + hiOff);
#pragma unroll
      for (int t = 0; t < 4; ++t) {
        const bf16* vRow = &vTile[cur][t * 16 + lane15][0];
        v16bf vF = load_frag(vRow + hiOff);
        o[t] = __builtin_amdgcn_wmma_f32_16x16x32_bf16(false, pF, false, vF, (short)0, o[t], false, false);
      }
    }
    __syncthreads();    // protect cur (to be overwritten next iteration)
  }

  // ---- normalize, write bf16 attn in [B, S, d_model] layout ----
#pragma unroll
  for (int t = 0; t < 4; ++t)
#pragma unroll
    for (int r = 0; r < 8; ++r) {
      const int s = q0 + r + hi8;
      float val = o[t][r] / lSt[r];
      Ob[(size_t)(b * SEQ + s) * D_MODEL + h * D_K + t * 16 + lane15] = to_bf16(val);
    }
}

// ---------- host launch ----------

extern "C" void kernel_launch(void* const* d_in, const int* in_sizes, int n_in,
                              void* d_out, int out_size, void* d_ws, size_t ws_size,
                              hipStream_t stream) {
  (void)in_sizes; (void)n_in; (void)out_size; (void)ws_size;
  const float* x  = (const float*)d_in[0];
  const float* Wq = (const float*)d_in[1];
  const float* Wk = (const float*)d_in[2];
  const float* Wv = (const float*)d_in[3];
  const float* Wo = (const float*)d_in[4];
  const int*   tp = (const int*)d_in[5];

  char* w = (char*)d_ws;
  bf16* xb  = (bf16*)w; w += (size_t)M_TOTAL * D_MODEL * 2;
  bf16* wqb = (bf16*)w; w += (size_t)D_MODEL * D_MODEL * 2;
  bf16* wkb = (bf16*)w; w += (size_t)D_MODEL * D_MODEL * 2;
  bf16* wvb = (bf16*)w; w += (size_t)D_MODEL * D_MODEL * 2;
  bf16* wob = (bf16*)w; w += (size_t)D_MODEL * D_MODEL * 2;
  bf16* qb  = (bf16*)w; w += (size_t)BATCH * NUM_HEADS * SEQ * D_K * 2;
  bf16* kb  = (bf16*)w; w += (size_t)BATCH * NUM_HEADS * SEQ * D_K * 2;
  bf16* vtb = (bf16*)w; w += (size_t)BATCH * NUM_HEADS * D_K * SEQ * 2;
  bf16* atb = (bf16*)w; w += (size_t)M_TOTAL * D_MODEL * 2;

  const int xN4 = M_TOTAL * D_MODEL / 4;
  const int wN4 = D_MODEL * D_MODEL / 4;
  cvt_f32_to_bf16<<<(xN4 + 255) / 256, 256, 0, stream>>>(x,  xb,  xN4);
  cvt_f32_to_bf16<<<(wN4 + 255) / 256, 256, 0, stream>>>(Wq, wqb, wN4);
  cvt_f32_to_bf16<<<(wN4 + 255) / 256, 256, 0, stream>>>(Wk, wkb, wN4);
  cvt_f32_to_bf16<<<(wN4 + 255) / 256, 256, 0, stream>>>(Wv, wvb, wN4);
  cvt_f32_to_bf16<<<(wN4 + 255) / 256, 256, 0, stream>>>(Wo, wob, wN4);

  dim3 gGemm(M_TOTAL / 128, D_MODEL / 64);   // 4 waves x 32 rows per block
  gemm16_kernel<0><<<gGemm, 128, 0, stream>>>(xb, wqb, qb,  tp);  // Q + RoPE
  gemm16_kernel<0><<<gGemm, 128, 0, stream>>>(xb, wkb, kb,  tp);  // K + RoPE
  gemm16_kernel<1><<<gGemm, 128, 0, stream>>>(xb, wvb, vtb, tp);  // V transposed

  attn_kernel<<<dim3(SEQ / 64, NUM_HEADS, BATCH), 128, 0, stream>>>(qb, kb, vtb, atb);

  gemm16_kernel<2><<<gGemm, 128, 0, stream>>>(atb, wob, d_out, tp);  // output proj (f32)
}
